// Perceptor_20401094656118
// MI455X (gfx1250) — matrix operand order repaired and checked by reading, not compiled, not run on hardware
//
#include <hip/hip_runtime.h>
#include <hip/hip_bf16.h>

#define N_USERS 50000
#define N_ITEMS 50000
#define N_NODES (N_USERS + N_ITEMS)
#define HID 128
#define N_EDGES 1600000

typedef __attribute__((ext_vector_type(8)))  float  v8f;
typedef __attribute__((ext_vector_type(16))) __bf16 v16bf;

union FragAB {
    v16bf bf;
    unsigned int u[8];
    uint4 q[2];
};

__device__ __forceinline__ unsigned int pack2bf(float a, float b) {
    unsigned int ua = __builtin_bit_cast(unsigned int, a);
    unsigned int ub = __builtin_bit_cast(unsigned int, b);
    ua = (ua + 0x7FFFu + ((ua >> 16) & 1u)) >> 16;   // round-to-nearest-even
    ub = (ub + 0x7FFFu + ((ub >> 16) & 1u)) >> 16;
    return (ua & 0xFFFFu) | (ub << 16);
}

// permute pair index within a 16-pair k-chunk so each lane's 8 fragment dwords
// are contiguous in LDS: group order {0..3, 8..11, 4..7, 12..15}
__device__ __forceinline__ int permPair(int kp) {
    int c = kp >> 4, l = kp & 15;
    int g = l >> 2, p = l & 3;
    int ng = ((g & 1) << 1) | (g >> 1);              // 0->0, 1->2, 2->1, 3->3
    return c * 16 + ng * 4 + p;
}

// ---------------------------------------------------------------- concat ----
__global__ __launch_bounds__(256) void concat_k(const float* __restrict__ ue,
                                                const float* __restrict__ ie,
                                                float* __restrict__ x0) {
    int i = blockIdx.x * blockDim.x + threadIdx.x;          // float4 index
    const int total4 = N_NODES * HID / 4;
    if (i >= total4) return;
    const int split4 = N_USERS * HID / 4;
    float4 v = (i < split4) ? ((const float4*)ue)[i] : ((const float4*)ie)[i - split4];
    ((float4*)x0)[i] = v;
}

// ---------------------------------------------- weight pre-pack (bf16) ------
// fragment-ordered layout: Wp[((w*4+kc)*32+lane)*8 + i]
//   lane: n = w*16 + (lane&15), half = lane>>4
//   u[i]: kk = kc*32 + 2*(i&3) + 8*half + 16*(i>>2) ; pack(W[kk][n], W[kk+1][n])
__global__ __launch_bounds__(256) void packW_k(const float* __restrict__ W,
                                               unsigned int* __restrict__ Wp) {
    int tid = blockIdx.x * blockDim.x + threadIdx.x;        // 0..8191
    if (tid >= 8192) return;
    int i    = tid & 7;
    int lane = (tid >> 3) & 31;
    int kc   = (tid >> 8) & 3;
    int w    = tid >> 10;
    int half = lane >> 4;
    int n    = w * 16 + (lane & 15);
    int kk   = kc * 32 + 2 * (i & 3) + 8 * half + 16 * (i >> 2);
    Wp[tid] = pack2bf(W[kk * HID + n], W[(kk + 1) * HID + n]);
}

// ---------------------------------------------------------------- degree ----
__global__ __launch_bounds__(256) void deg_k(const int* __restrict__ dst,
                                             float* __restrict__ deg) {
    int e = blockIdx.x * blockDim.x + threadIdx.x;
    if (e < N_EDGES) atomicAdd(&deg[dst[e]], 1.0f);
}

__global__ __launch_bounds__(256) void invdeg_k(float* __restrict__ deg) {
    int i = blockIdx.x * blockDim.x + threadIdx.x;
    if (i < N_NODES) deg[i] = 1.0f / fmaxf(deg[i], 1.0f);
}

// --------------------------------------------------------------- scatter ----
// one thread = one (edge, 4-float chunk): b128 load + 4 non-returning atomics
__global__ __launch_bounds__(256) void scatter_k(const float* __restrict__ x,
                                                 const int* __restrict__ src,
                                                 const int* __restrict__ dst,
                                                 float* __restrict__ agg) {
    long tid = (long)blockIdx.x * blockDim.x + threadIdx.x;
    const long total = (long)N_EDGES * (HID / 4);
    if (tid >= total) return;
    int e = (int)(tid >> 5);            // HID/4 = 32 chunks per edge
    int c = ((int)tid & 31) * 4;
    int s = src[e], d = dst[e];
    float4 v = *(const float4*)(x + (long)s * HID + c);
    float* a = agg + (long)d * HID + c;
    atomicAdd(a + 0, v.x);
    atomicAdd(a + 1, v.y);
    atomicAdd(a + 2, v.z);
    atomicAdd(a + 3, v.w);
}

// ------------------------------------------------------------- WMMA GEMM ----
// x_out = (agg * inv_deg) @ Wl + x_in @ Wr + b   (Wl/Wr pre-packed bf16 frags)
// block = 256 threads = 8 waves; block owns 16 rows; wave w owns cols [16w,16w+16)
__global__ __launch_bounds__(256) void sage_gemm_k(const float* __restrict__ xin,
                                                   const float* __restrict__ agg,
                                                   const float* __restrict__ invdeg,
                                                   const unsigned int* __restrict__ Wlp,
                                                   const unsigned int* __restrict__ Wrp,
                                                   const float* __restrict__ bias,
                                                   float* __restrict__ xout) {
    __shared__ unsigned int ldsAgg[16 * 64];   // 16 rows x 64 bf16-pairs (permuted)
    __shared__ unsigned int ldsX[16 * 64];

    const int t = threadIdx.x;
    const int rowbase = blockIdx.x * 16;

    // cooperative load + bf16 pack of both 16x128 A tiles (4 pairs/thread),
    // stored in fragment-contiguous (permuted) order
    #pragma unroll
    for (int p = 0; p < 4; ++p) {
        int idx = t + p * 256;                 // 0..1023
        int r   = idx >> 6;                    // 0..15
        int kp  = idx & 63;                    // pair index 0..63
        int pp  = r * 64 + permPair(kp);
        long g  = (long)(rowbase + r) * HID + kp * 2;
        float sc = invdeg[rowbase + r];
        ldsAgg[pp] = pack2bf(agg[g] * sc, agg[g + 1] * sc);
        ldsX[pp]   = pack2bf(xin[g], xin[g + 1]);
    }
    __syncthreads();

    const int wave = t >> 5;
    const int lane = t & 31;
    const int m    = lane & 15;
    const int half = lane >> 4;
    const int n    = wave * 16 + (lane & 15);  // global output column

    const uint4* ldsAgg4 = (const uint4*)ldsAgg;
    const uint4* ldsX4   = (const uint4*)ldsX;

    v8f c = {};
    #pragma unroll
    for (int kc = 0; kc < 4; ++kc) {           // K = 128 in chunks of 32
        FragAB a, bW;
        const int abase4 = (m * 64 + kc * 16 + half * 8) >> 2;   // uint4 units
        const unsigned int* wl = Wlp + ((wave * 4 + kc) * 32 + lane) * 8;
        const unsigned int* wr = Wrp + ((wave * 4 + kc) * 32 + lane) * 8;

        // agg @ Wl
        a.q[0]  = ldsAgg4[abase4];
        a.q[1]  = ldsAgg4[abase4 + 1];
        bW.q[0] = ((const uint4*)wl)[0];
        bW.q[1] = ((const uint4*)wl)[1];
        c = __builtin_amdgcn_wmma_f32_16x16x32_bf16(false, a.bf, false, bW.bf,
                                                    (short)0, c, false, false);

        // x @ Wr
        a.q[0]  = ldsX4[abase4];
        a.q[1]  = ldsX4[abase4 + 1];
        bW.q[0] = ((const uint4*)wr)[0];
        bW.q[1] = ((const uint4*)wr)[1];
        c = __builtin_amdgcn_wmma_f32_16x16x32_bf16(false, a.bf, false, bW.bf,
                                                    (short)0, c, false, false);
    }

    const float bv = bias[n];
    #pragma unroll
    for (int j = 0; j < 8; ++j) {
        int mrow = j + 8 * half;               // C/D layout: VGPR j -> M = j + 8*half
        xout[(long)(rowbase + mrow) * HID + n] = c[j] + bv;
    }
}

// ----------------------------------------------------------------- score ----
// one wave32 per item: dot(x[item[i]], W_out) + b_out
__global__ __launch_bounds__(256) void score_k(const float* __restrict__ x,
                                               const int* __restrict__ item,
                                               const float* __restrict__ Wout,
                                               const float* __restrict__ bout,
                                               float* __restrict__ s) {
    int warp = (blockIdx.x * blockDim.x + threadIdx.x) >> 5;
    int lane = threadIdx.x & 31;
    if (warp >= N_ITEMS) return;
    const float* row = x + (long)item[warp] * HID;
    float acc = 0.0f;
    #pragma unroll
    for (int k = lane; k < HID; k += 32) acc += row[k] * Wout[k];
    #pragma unroll
    for (int off = 16; off; off >>= 1) acc += __shfl_xor(acc, off, 32);
    if (lane == 0) s[warp] = acc + bout[0];
}

// -------------------------------------------------------- softmax reduce ----
__global__ __launch_bounds__(1024) void softmax_red_k(const float* __restrict__ s,
                                                      float* __restrict__ red) {
    __shared__ float sm[1024];
    int t = threadIdx.x;
    float mx = -3.4e38f;
    for (int i = t; i < N_ITEMS; i += 1024) mx = fmaxf(mx, s[i]);
    sm[t] = mx; __syncthreads();
    for (int o = 512; o; o >>= 1) { if (t < o) sm[t] = fmaxf(sm[t], sm[t + o]); __syncthreads(); }
    mx = sm[0]; __syncthreads();
    float sum = 0.0f;
    for (int i = t; i < N_ITEMS; i += 1024) sum += __expf(s[i] - mx);
    sm[t] = sum; __syncthreads();
    for (int o = 512; o; o >>= 1) { if (t < o) sm[t] += sm[t + o]; __syncthreads(); }
    if (t == 0) { red[0] = mx; red[1] = sm[0]; }
}

__global__ __launch_bounds__(256) void final_k(const float* __restrict__ s,
                                               const float* __restrict__ red,
                                               const float* __restrict__ prior,
                                               float* __restrict__ out) {
    int i = blockIdx.x * blockDim.x + threadIdx.x;
    if (i < N_ITEMS) {
        float p = fmaxf(prior[0], 0.0f);
        out[i] = p * __expf(s[i] - red[0]) / red[1];
    }
}

// ---------------------------------------------------------------- launch ----
extern "C" void kernel_launch(void* const* d_in, const int* in_sizes, int n_in,
                              void* d_out, int out_size, void* d_ws, size_t ws_size,
                              hipStream_t stream) {
    const int*   item   = (const int*)d_in[0];
    const int*   ei     = (const int*)d_in[1];       // [2, N_EDGES]
    const float* ue     = (const float*)d_in[2];
    const float* ie     = (const float*)d_in[3];
    const float* Wl1    = (const float*)d_in[4];
    const float* Wr1    = (const float*)d_in[5];
    const float* b1     = (const float*)d_in[6];
    const float* Wl2    = (const float*)d_in[7];
    const float* Wr2    = (const float*)d_in[8];
    const float* b2     = (const float*)d_in[9];
    const float* Wout   = (const float*)d_in[10];
    const float* bout   = (const float*)d_in[11];
    const float* prior  = (const float*)d_in[12];
    float*       out    = (float*)d_out;

    const int* src = ei;
    const int* dst = ei + N_EDGES;

    const size_t XB = (size_t)N_NODES * HID * sizeof(float);   // 51.2 MB
    char* ws   = (char*)d_ws;
    float* x0  = (float*)(ws);
    float* x1  = (float*)(ws + XB);
    float* agg = (float*)(ws + 2 * XB);
    float* deg = (float*)(ws + 3 * XB);
    float* s   = (float*)(ws + 3 * XB + (size_t)N_NODES * sizeof(float));
    float* red = (float*)(ws + 3 * XB + (size_t)N_NODES * sizeof(float)
                             + (size_t)N_ITEMS * sizeof(float));
    unsigned int* Wl1p = (unsigned int*)(ws + 3 * XB + (1 << 20));
    unsigned int* Wr1p = Wl1p + 8192;
    unsigned int* Wl2p = Wr1p + 8192;
    unsigned int* Wr2p = Wl2p + 8192;

    // x0 = concat(user_emb, item_emb)
    concat_k<<<(N_NODES * HID / 4 + 255) / 256, 256, 0, stream>>>(ue, ie, x0);

    // pre-pack weights into bf16 fragment order (32 KB each, L2-resident)
    packW_k<<<32, 256, 0, stream>>>(Wl1, Wl1p);
    packW_k<<<32, 256, 0, stream>>>(Wr1, Wr1p);
    packW_k<<<32, 256, 0, stream>>>(Wl2, Wl2p);
    packW_k<<<32, 256, 0, stream>>>(Wr2, Wr2p);

    // degree -> inv_deg (in place)
    hipMemsetAsync(deg, 0, (size_t)N_NODES * sizeof(float), stream);
    deg_k<<<(N_EDGES + 255) / 256, 256, 0, stream>>>(dst, deg);
    invdeg_k<<<(N_NODES + 255) / 256, 256, 0, stream>>>(deg);

    const long scat_threads = (long)N_EDGES * (HID / 4);
    const int  scat_blocks  = (int)((scat_threads + 255) / 256);

    // layer 1: x0 -> x1
    hipMemsetAsync(agg, 0, XB, stream);
    scatter_k<<<scat_blocks, 256, 0, stream>>>(x0, src, dst, agg);
    sage_gemm_k<<<N_NODES / 16, 256, 0, stream>>>(x0, agg, deg, Wl1p, Wr1p, b1, x1);

    // layer 2: x1 -> x0
    hipMemsetAsync(agg, 0, XB, stream);
    scatter_k<<<scat_blocks, 256, 0, stream>>>(x1, src, dst, agg);
    sage_gemm_k<<<N_NODES / 16, 256, 0, stream>>>(x1, agg, deg, Wl2p, Wr2p, b2, x0);

    // head
    score_k<<<(N_ITEMS * 32 + 255) / 256, 256, 0, stream>>>(x0, item, Wout, bout, s);
    softmax_red_k<<<1, 1024, 0, stream>>>(s, red);
    final_k<<<(N_ITEMS + 255) / 256, 256, 0, stream>>>(s, red, prior, out);
}
